// HierarchicalGNN_11982958756500
// MI455X (gfx1250) — matrix operand order repaired and checked by reading, not compiled, not run on hardware
//
#include <hip/hip_runtime.h>
#include <cstring>

// ---------------------------------------------------------------------------
// CDNA5 (gfx1250) hierarchical-GNN forward.  All GEMMs (K=128) run on
// v_wmma_f32_16x16x32_f16 (f16 operands, f32 accumulate).  Pooling softmax
// exploits the exact-zero structure of the -1e9 mask and stages score rows
// into LDS via global_load_async_to_lds_b128 (ASYNCcnt path).  Wave32.
// ---------------------------------------------------------------------------

typedef __attribute__((ext_vector_type(16))) _Float16 v16h;
typedef __attribute__((ext_vector_type(8)))  float    v8f;

#define DEV __device__ __forceinline__

// ---- problem constants ----
#define BN    16
#define L1N_  4096
#define L2N_  512
#define L3N_  64
#define HD    128
#define L1E_  65536
#define L2E_  8192
#define N1_   (BN*L1N_)          // 65536
#define N2_   (BN*L2N_)          // 8192
#define BE1_  (BN*L1E_)          // 1048576
#define BE2_  (BN*L2E_)          // 131072
#define E2T_  (BE2_ + N2_)       // 139264

DEV float wsum(float v) {
#pragma unroll
  for (int m = 16; m > 0; m >>= 1) v += __shfl_xor(v, m, 32);
  return v;
}

DEV unsigned fordered(float f) {
  unsigned u = __float_as_uint(f);
  return (u >> 31) ? ~u : (u | 0x80000000u);
}
DEV float funordered(unsigned o) {
  unsigned u = (o >> 31) ? (o & 0x7FFFFFFFu) : ~o;
  return __uint_as_float(u);
}

// Async global->LDS copy (16B per lane), tracked by ASYNCcnt.
// Flat pointers to __shared__ have their LDS byte offset in the low 32 bits
DEV void async_copy_b128(float* lds_dst, const float* gsrc) {
  unsigned off = (unsigned)(size_t)lds_dst;
  asm volatile("global_load_async_to_lds_b128 %0, %1, off"
               :: "v"(off), "v"(gsrc)
               : "memory");
}
DEV void wait_async0() { asm volatile("s_wait_asynccnt 0x0" ::: "memory"); }

// Stage a 16x128 f32 tile -> f16 LDS tile (pitch 136 halves, 16B-aligned rows).
// Exactly 256 threads: 16 threads/row, 8 floats each (two float4 loads,
// one packed 16-byte LDS store).
DEV void stage_a16(const float* __restrict__ A, _Float16* __restrict__ As, int tid) {
  int m = tid >> 4;
  int k = (tid & 15) * 8;
  const float4* src = (const float4*)(A + (size_t)m * 128 + k);
  float4 a = src[0], b = src[1];
  union { _Float16 h[8]; uint4 q; } u;
  u.h[0] = (_Float16)a.x; u.h[1] = (_Float16)a.y;
  u.h[2] = (_Float16)a.z; u.h[3] = (_Float16)a.w;
  u.h[4] = (_Float16)b.x; u.h[5] = (_Float16)b.y;
  u.h[6] = (_Float16)b.z; u.h[7] = (_Float16)b.w;
  *(uint4*)(As + m * 136 + k) = u.q;
}

// ---------------------------------------------------------------------------
// small utility kernels
// ---------------------------------------------------------------------------
__global__ void fill_f32(float* p, float v, int n) {
  int i = blockIdx.x * 256 + threadIdx.x;
  if (i < n) p[i] = v;
}
__global__ void fill_u32(unsigned* p, unsigned v, int n) {
  int i = blockIdx.x * 256 + threadIdx.x;
  if (i < n) p[i] = v;
}
// W[128][N] f32 row-major -> Wt[N][128] f16 (B-fragment friendly)
__global__ void conv_wt(const float* __restrict__ W, _Float16* __restrict__ Wt, int N) {
  int i = blockIdx.x * 256 + threadIdx.x;
  if (i < N * 128) {
    int n = i >> 7, k = i & 127;
    Wt[(size_t)n * 128 + k] = (_Float16)W[(size_t)k * N + n];
  }
}

// ---------------------------------------------------------------------------
// GCN level-1 pieces
// ---------------------------------------------------------------------------
__global__ void gcn_deg(const int* __restrict__ e1, float* __restrict__ deg) {
  int tid = blockIdx.x * 256 + threadIdx.x;            // exactly BE1_
  int b = tid >> 16, k = tid & 65535;
  int d = e1[L1E_ + k] + b * L1N_;
  atomicAdd(&deg[d], 1.0f);
}
__global__ void deg2dinv(float* p, int n) {
  int i = blockIdx.x * 256 + threadIdx.x;
  if (i < n) p[i] = rsqrtf(fmaxf(p[i], 1e-12f));
}
__global__ void gcn1_h(const float* __restrict__ x, const float* __restrict__ W1,
                       float* __restrict__ h) {
  int i = blockIdx.x * 256 + threadIdx.x;              // N1_*HD
  int node = i >> 7, c = i & 127;
  h[i] = x[node] * W1[c];
}
// self-loop term: agg = dinv^2 * h
__global__ void gcn_self(const float* __restrict__ h, const float* __restrict__ dinv,
                         float* __restrict__ agg) {
  int i = blockIdx.x * 256 + threadIdx.x;              // N1_*HD
  int node = i >> 7;
  float dv = dinv[node];
  agg[i] = dv * dv * h[i];
}
__global__ void gcn_edge(const int* __restrict__ e1, const float* __restrict__ h,
                         const float* __restrict__ dinv, float* __restrict__ agg) {
  int tid = blockIdx.x * 256 + threadIdx.x;            // BE1_*128
  int hc = tid & 127;
  int eb = tid >> 7;
  int b = eb >> 16, k = eb & 65535;
  int s = e1[k] + b * L1N_;
  int d = e1[L1E_ + k] + b * L1N_;
  atomicAdd(&agg[(size_t)d * HD + hc], h[(size_t)s * HD + hc] * dinv[s] * dinv[d]);
}

// LayerNorm over H=128, optional residual.  One wave per row.
__global__ __launch_bounds__(256) void ln128(const float* __restrict__ in,
                                             const float* __restrict__ bias,
                                             const float* __restrict__ gam,
                                             const float* __restrict__ bet,
                                             const float* __restrict__ resid,
                                             float* __restrict__ outp) {
  int row  = blockIdx.x * 8 + (threadIdx.x >> 5);
  int lane = threadIdx.x & 31;
  size_t off = (size_t)row * HD + lane * 4;
  float4 v  = *(const float4*)(in + off);
  float4 bb = *(const float4*)(bias + lane * 4);
  float a0 = v.x + bb.x, a1 = v.y + bb.y, a2 = v.z + bb.z, a3 = v.w + bb.w;
  float mean = wsum(a0 + a1 + a2 + a3) * (1.0f / 128.0f);
  float d0 = a0 - mean, d1 = a1 - mean, d2 = a2 - mean, d3 = a3 - mean;
  float var = wsum(d0 * d0 + d1 * d1 + d2 * d2 + d3 * d3) * (1.0f / 128.0f);
  float rs = rsqrtf(var + 1e-5f);
  float4 g4 = *(const float4*)(gam + lane * 4);
  float4 b4 = *(const float4*)(bet + lane * 4);
  float r0 = d0 * rs * g4.x + b4.x;
  float r1 = d1 * rs * g4.y + b4.y;
  float r2 = d2 * rs * g4.z + b4.z;
  float r3 = d3 * rs * g4.w + b4.w;
  if (resid) {
    float4 rr = *(const float4*)(resid + off);
    r0 += rr.x; r1 += rr.y; r2 += rr.z; r3 += rr.w;
  }
  float4 o; o.x = r0; o.y = r1; o.z = r2; o.w = r3;
  *(float4*)(outp + off) = o;
}

// ---------------------------------------------------------------------------
// WMMA GEMM, N fixed at 128:  C[M,128] = A[M,128] * W[128,128]
// (Wt = W^T as f16 [128][128]).  block: 256 thr (8 waves); grid: M/16;
// wave w owns n-tile w.  Compile-time N folds C-store strides into offsets.
// ---------------------------------------------------------------------------
__global__ __launch_bounds__(256) void gemm128(const float* __restrict__ A,
                                               const _Float16* __restrict__ Bt,
                                               float* __restrict__ C) {
  __shared__ _Float16 As[16 * 136];
  const int m0 = blockIdx.x << 4;
  stage_a16(A + (size_t)m0 * 128, As, threadIdx.x);
  __builtin_prefetch(A + (size_t)(m0 + 16) * 128 + threadIdx.x * 8, 0, 0);
  __syncthreads();
  const int wave = threadIdx.x >> 5;
  const int lane = threadIdx.x & 31;
  const int g = lane >> 4, ln = lane & 15;
  const int n0 = wave * 16;
  union { v16h v; uint4 q[2]; } af, bf;
  v8f c = {};
#pragma unroll
  for (int k0 = 0; k0 < 128; k0 += 32) {
    af.q[0] = *(const uint4*)(As + ln * 136 + k0 + g * 8);        // halves 0-7
    af.q[1] = *(const uint4*)(As + ln * 136 + k0 + 16 + g * 8);   // halves 8-15
    const _Float16* bp = Bt + (size_t)(n0 + ln) * 128 + k0 + g * 8;
    bf.q[0] = *(const uint4*)bp; bf.q[1] = *(const uint4*)(bp + 16);
    c = __builtin_amdgcn_wmma_f32_16x16x32_f16(false, af.v, false, bf.v,
                                               (short)0, c, false, false);
  }
  float* cp = C + (size_t)(m0 + g * 8) * 128 + n0 + ln;
#pragma unroll
  for (int j = 0; j < 8; ++j)
    cp[(size_t)j * 128] = c[j];
}

// ---------------------------------------------------------------------------
// Fused Source2Token score: s = fc2(tanh(LN(X @ W1 + b1))) + b2
// Block computes full 16x512 fc1 tile via WMMA into LDS, then per-row
// LN/tanh/dot without ever touching HBM with the 512-wide intermediate.
// ---------------------------------------------------------------------------
__global__ __launch_bounds__(256) void s2t_score(const float* __restrict__ X,
                                                 const _Float16* __restrict__ Wt,
                                                 const float* __restrict__ b1,
                                                 const float* __restrict__ lng,
                                                 const float* __restrict__ lnb,
                                                 const float* __restrict__ w2,
                                                 const float* __restrict__ b2,
                                                 float* __restrict__ sout) {
  __shared__ _Float16 As[16 * 136];
  __shared__ float Cs[16 * 520];
  const int m0 = blockIdx.x << 4;
  stage_a16(X + (size_t)m0 * 128, As, threadIdx.x);
  __syncthreads();
  const int wave = threadIdx.x >> 5;
  const int lane = threadIdx.x & 31;
  const int g = lane >> 4, ln = lane & 15;
#pragma unroll
  for (int i = 0; i < 4; ++i) {                        // 8 waves x 4 tiles = N=512
    const int n0 = (wave * 4 + i) * 16;
    union { v16h v; uint4 q[2]; } af, bf;
    v8f c = {};
#pragma unroll
    for (int k0 = 0; k0 < 128; k0 += 32) {
      af.q[0] = *(const uint4*)(As + ln * 136 + k0 + g * 8);
      af.q[1] = *(const uint4*)(As + ln * 136 + k0 + 16 + g * 8);
      const _Float16* bp = Wt + (size_t)(n0 + ln) * 128 + k0 + g * 8;
      bf.q[0] = *(const uint4*)bp; bf.q[1] = *(const uint4*)(bp + 16);
      c = __builtin_amdgcn_wmma_f32_16x16x32_f16(false, af.v, false, bf.v,
                                                 (short)0, c, false, false);
    }
#pragma unroll
    for (int j = 0; j < 8; ++j)
      Cs[(j + g * 8) * 520 + n0 + ln] = c[j];
  }
  __syncthreads();
  // epilogue: wave handles rows 2w, 2w+1 ; LN over 512, tanh, dot with w2
  for (int rr = 0; rr < 2; ++rr) {
    const int r = wave * 2 + rr;
    float* row = Cs + r * 520;
    float sum = 0.f;
    for (int n = lane; n < 512; n += 32) { float v = row[n] + b1[n]; row[n] = v; sum += v; }
    float mean = wsum(sum) * (1.0f / 512.0f);
    float vs = 0.f;
    for (int n = lane; n < 512; n += 32) { float d = row[n] - mean; vs += d * d; }
    float rstd = rsqrtf(wsum(vs) * (1.0f / 512.0f) + 1e-5f);
    float acc = 0.f;
    for (int n = lane; n < 512; n += 32) {
      float z = tanhf((row[n] - mean) * rstd * lng[n] + lnb[n]);
      acc += z * w2[n];
    }
    acc = wsum(acc);
    if (lane == 0) sout[m0 + r] = acc + b2[0];
  }
}

// ---------------------------------------------------------------------------
// Masked-softmax pooling: out[b,g,:] = softmax_n(mask? s : -1e9) @ X
// Unmasked weights underflow to exact 0 (same as reference), so only masked
// nodes contribute.  One block per (g,b); thread t owns channel h=t.
// Score row is staged into LDS with async global->LDS copies (ASYNCcnt).
// ---------------------------------------------------------------------------
__global__ __launch_bounds__(128) void s2t_pool(const float* __restrict__ X,
                                                const float* __restrict__ s,
                                                const int* __restrict__ ind,
                                                float* __restrict__ outp, int Nn) {
  __shared__ float sraw[4096];
  __shared__ float wsh[4096];
  __shared__ float red[128];
  const int g = blockIdx.x, b = blockIdx.y, G = gridDim.x;
  const int t = threadIdx.x;
  const int base = b * Nn;
  const int* irow = ind + (size_t)g * Nn;
  // async-stage this graph's score row (Nn f32, 16B per lane per issue)
  const float* sp = s + base;
  for (int c4 = t; c4 < (Nn >> 2); c4 += 128)
    async_copy_b128(&sraw[c4 * 4], sp + c4 * 4);
  wait_async0();
  __syncthreads();

  float mx = -1e9f;
  for (int n = t; n < Nn; n += 128) {
    float v = irow[n] ? sraw[n] : -1e9f;
    wsh[n] = v;
    mx = fmaxf(mx, v);
  }
  red[t] = mx; __syncthreads();
  for (int o = 64; o > 0; o >>= 1) { if (t < o) red[t] = fmaxf(red[t], red[t + o]); __syncthreads(); }
  const float MX = red[0]; __syncthreads();
  float zs = 0.f;
  for (int n = t; n < Nn; n += 128) {
    float w = expf(wsh[n] - MX);                       // -1e9 entries -> exactly 0
    wsh[n] = w; zs += w;
  }
  red[t] = zs; __syncthreads();
  for (int o = 64; o > 0; o >>= 1) { if (t < o) red[t] += red[t + o]; __syncthreads(); }
  const float Z = red[0]; __syncthreads();
  float acc = 0.f;
  const int h = t;
  for (int n = 0; n < Nn; ++n) {
    float w = wsh[n];
    if (w > 0.0f) acc += w * X[(size_t)(base + n) * HD + h];
  }
  outp[(size_t)(b * G + g) * HD + h] = acc / Z;
}

// ---------------------------------------------------------------------------
// GAT pieces (8192 nodes, 139264 edges incl. self loops)
// ---------------------------------------------------------------------------
DEV void edge2sd(const int* e2, int eb, int& s, int& d) {
  if (eb < BE2_) {
    int b = eb >> 13, k = eb & 8191;
    s = e2[k] + b * L2N_;
    d = e2[L2E_ + k] + b * L2N_;
  } else {
    s = d = eb - BE2_;
  }
}
__global__ __launch_bounds__(256) void gat_asad(const float* __restrict__ hg,
                                                const float* __restrict__ a_s,
                                                const float* __restrict__ a_d,
                                                float* __restrict__ asv,
                                                float* __restrict__ adv) {
  int row  = blockIdx.x * 8 + (threadIdx.x >> 5);
  int lane = threadIdx.x & 31;
  float4 h4 = *(const float4*)(hg + (size_t)row * HD + lane * 4);
  float4 s4 = *(const float4*)(a_s + lane * 4);
  float4 d4 = *(const float4*)(a_d + lane * 4);
  float ds = wsum(h4.x * s4.x + h4.y * s4.y + h4.z * s4.z + h4.w * s4.w);
  float dd = wsum(h4.x * d4.x + h4.y * d4.y + h4.z * d4.z + h4.w * d4.w);
  if (lane == 0) { asv[row] = ds; adv[row] = dd; }
}
__global__ void gat_logit_max(const int* __restrict__ e2, const float* __restrict__ asv,
                              const float* __restrict__ adv, float* __restrict__ lg,
                              unsigned* __restrict__ mx) {
  int eb = blockIdx.x * 256 + threadIdx.x;             // exactly E2T_
  int s, d; edge2sd(e2, eb, s, d);
  float x = asv[s] + adv[d];
  float l = x > 0.f ? x : 0.2f * x;                    // leaky_relu 0.2
  lg[eb] = l;
  atomicMax(&mx[d], fordered(l));
}
__global__ void gat_exp(const int* __restrict__ e2, float* __restrict__ ew,
                        const unsigned* __restrict__ mx, float* __restrict__ z) {
  int eb = blockIdx.x * 256 + threadIdx.x;
  int s, d; edge2sd(e2, eb, s, d);
  float w = expf(ew[eb] - funordered(mx[d]));
  ew[eb] = w;
  atomicAdd(&z[d], w);
}
__global__ void gat_agg(const int* __restrict__ e2, const float* __restrict__ ew,
                        const float* __restrict__ z, const float* __restrict__ hg,
                        float* __restrict__ agg) {
  int tid = blockIdx.x * 256 + threadIdx.x;            // E2T_*128
  int h = tid & 127, eb = tid >> 7;
  int s, d; edge2sd(e2, eb, s, d);
  atomicAdd(&agg[(size_t)d * HD + h], (ew[eb] / z[d]) * hg[(size_t)s * HD + h]);
}

// final gather + concat -> [B, 64*256]
__global__ void concat_out(const float* __restrict__ L1B, const float* __restrict__ lv3,
                           const int* __restrict__ idx13, float* __restrict__ out) {
  int tid = blockIdx.x * 256 + threadIdx.x;            // 262144
  int b = tid >> 14, rem = tid & 16383;
  int t = rem >> 8, c = rem & 255;
  out[tid] = (c < 128)
                 ? L1B[((size_t)b * L1N_ + idx13[t]) * HD + c]
                 : lv3[((size_t)(b * L3N_ + t)) * HD + (c - 128)];
}

// ---------------------------------------------------------------------------
extern "C" void kernel_launch(void* const* d_in, const int* in_sizes, int n_in,
                              void* d_out, int out_size, void* d_ws, size_t ws_size,
                              hipStream_t stream) {
  (void)in_sizes; (void)n_in; (void)out_size; (void)ws_size;
  const float* x      = (const float*)d_in[0];
  const int*   e1     = (const int*)d_in[1];
  const int*   e2     = (const int*)d_in[2];
  const int*   ind21  = (const int*)d_in[3];
  const int*   ind32  = (const int*)d_in[4];
  const int*   idx13  = (const int*)d_in[5];
  const float* W1     = (const float*)d_in[6];
  const float* b1c    = (const float*)d_in[7];
  const float* ln1_g  = (const float*)d_in[8];
  const float* ln1_b  = (const float*)d_in[9];
  const float* W2     = (const float*)d_in[10];
  const float* b2c    = (const float*)d_in[11];
  const float* ln2_g  = (const float*)d_in[12];
  const float* ln2_b  = (const float*)d_in[13];
  const float* Wg     = (const float*)d_in[14];
  const float* bgc    = (const float*)d_in[15];
  const float* a_src  = (const float*)d_in[16];
  const float* a_dst  = (const float*)d_in[17];
  const float* lng_g  = (const float*)d_in[18];
  const float* lng_b  = (const float*)d_in[19];
  const float* s12_w1 = (const float*)d_in[20];
  const float* s12_b1 = (const float*)d_in[21];
  const float* s12_lng= (const float*)d_in[22];
  const float* s12_lnb= (const float*)d_in[23];
  const float* s12_w2 = (const float*)d_in[24];
  const float* s12_b2 = (const float*)d_in[25];
  const float* s23_w1 = (const float*)d_in[26];
  const float* s23_b1 = (const float*)d_in[27];
  const float* s23_lng= (const float*)d_in[28];
  const float* s23_lnb= (const float*)d_in[29];
  const float* s23_w2 = (const float*)d_in[30];
  const float* s23_b2 = (const float*)d_in[31];

  // ---- workspace layout (floats) ----
  float* ws   = (float*)d_ws;
  float* TMP1 = ws;                       // 8,388,608  (h buffers; later level-2 arrays)
  float* TMP2 = ws + 8388608;             // 8,388,608  (agg buffers; later f16 W / GAT scratch)
  float* L1A  = ws + 16777216;            // lv1 after conv block 1
  float* L1B  = ws + 25165824;            // lv1 after conv block 2 (live till end)
  float* SMB  = ws + 33554432;
  float* dinv1 = SMB;                     // 65536 (deg -> dinv)
  float* s1    = SMB + 65536;             // 65536 level-1 scores
  float* s2    = SMB + 131072;            // 8192 level-2 scores
  _Float16* W2t = (_Float16*)(SMB + 139264);   // 128x128 f16
  _Float16* Wgt = (_Float16*)(SMB + 147456);   // 128x128 f16
  // reuse TMP1 after L1B is finalized:
  float* lv2  = TMP1;                     // 8192x128
  float* hg   = TMP1 + 1048576;           // 8192x128
  float* aggG = TMP1 + 2097152;           // 8192x128
  float* lv2b = TMP1 + 3145728;           // 8192x128
  float* lv3  = TMP1 + 4194304;           // 16x64x128
  // reuse TMP2 after L1B is finalized:
  _Float16* W12t = (_Float16*)TMP2;            // 512x128 f16
  _Float16* W23t = (_Float16*)(TMP2 + 32768);  // 512x128 f16
  float*    asv  = TMP2 + 65536;          // 8192
  float*    adv  = TMP2 + 73728;          // 8192
  unsigned* mx2  = (unsigned*)(TMP2 + 81920);  // 8192 (ordered uints)
  float*    z2   = TMP2 + 90112;          // 8192
  float*    ew   = TMP2 + 98304;          // 139264 (logits, then exp weights)

  float negBig = -1e30f; unsigned ub; memcpy(&ub, &negBig, 4);
  const unsigned ordInit = ~ub;           // fordered(-1e30f)

  // ---- GCN level 1, block 1 ----
  fill_f32<<<N1_ / 256, 256, 0, stream>>>(dinv1, 1.0f, N1_);        // self-loop deg
  gcn_deg<<<BE1_ / 256, 256, 0, stream>>>(e1, dinv1);
  deg2dinv<<<N1_ / 256, 256, 0, stream>>>(dinv1, N1_);
  gcn1_h<<<(N1_ * HD) / 256, 256, 0, stream>>>(x, W1, TMP1);
  gcn_self<<<(N1_ * HD) / 256, 256, 0, stream>>>(TMP1, dinv1, TMP2);
  gcn_edge<<<(BE1_ * 128) / 256, 256, 0, stream>>>(e1, TMP1, dinv1, TMP2);
  ln128<<<N1_ / 8, 256, 0, stream>>>(TMP2, b1c, ln1_g, ln1_b, nullptr, L1A);

  // ---- GCN level 1, block 2 (WMMA GEMM) + residual ----
  conv_wt<<<64, 256, 0, stream>>>(W2, W2t, 128);
  gemm128<<<N1_ / 16, 256, 0, stream>>>(L1A, W2t, TMP1);
  gcn_self<<<(N1_ * HD) / 256, 256, 0, stream>>>(TMP1, dinv1, TMP2);
  gcn_edge<<<(BE1_ * 128) / 256, 256, 0, stream>>>(e1, TMP1, dinv1, TMP2);
  ln128<<<N1_ / 8, 256, 0, stream>>>(TMP2, b2c, ln2_g, ln2_b, L1A, L1B);

  // ---- Source2Token pooling 4096 -> 512 ----
  conv_wt<<<256, 256, 0, stream>>>(s12_w1, W12t, 512);
  s2t_score<<<N1_ / 16, 256, 0, stream>>>(L1B, W12t, s12_b1, s12_lng, s12_lnb,
                                          s12_w2, s12_b2, s1);
  s2t_pool<<<dim3(L2N_, BN), 128, 0, stream>>>(L1B, s1, ind21, lv2, L1N_);

  // ---- GAT on level 2 ----
  conv_wt<<<64, 256, 0, stream>>>(Wg, Wgt, 128);
  gemm128<<<N2_ / 16, 256, 0, stream>>>(lv2, Wgt, hg);
  gat_asad<<<N2_ / 8, 256, 0, stream>>>(hg, a_src, a_dst, asv, adv);
  fill_u32<<<N2_ / 256, 256, 0, stream>>>(mx2, ordInit, N2_);
  fill_f32<<<N2_ / 256, 256, 0, stream>>>(z2, 0.0f, N2_);
  gat_logit_max<<<E2T_ / 256, 256, 0, stream>>>(e2, asv, adv, ew, mx2);
  gat_exp<<<E2T_ / 256, 256, 0, stream>>>(e2, ew, mx2, z2);
  fill_f32<<<(N2_ * HD) / 256, 256, 0, stream>>>(aggG, 0.0f, N2_ * HD);
  gat_agg<<<(E2T_ * 128) / 256, 256, 0, stream>>>(e2, ew, z2, hg, aggG);
  ln128<<<N2_ / 8, 256, 0, stream>>>(aggG, bgc, lng_g, lng_b, nullptr, lv2b);

  // ---- Source2Token pooling 512 -> 64 ----
  conv_wt<<<256, 256, 0, stream>>>(s23_w1, W23t, 512);
  s2t_score<<<N2_ / 16, 256, 0, stream>>>(lv2b, W23t, s23_b1, s23_lng, s23_lnb,
                                          s23_w2, s23_b2, s2);
  s2t_pool<<<dim3(L3N_, BN), 128, 0, stream>>>(lv2b, s2, ind32, lv3, L2N_);

  // ---- gather level-1 nodes + concat ----
  concat_out<<<(BN * L3N_ * 256) / 256, 256, 0, stream>>>(L1B, lv3, idx13,
                                                          (float*)d_out);
}
